// GuidedGNN_28235115004306
// MI455X (gfx1250) — compile-verified, matrix operand order checked by reading, not run on hardware
//
#include <hip/hip_runtime.h>
#include <math.h>

// ---------------------------------------------------------------------------
// Hetero-GAT (2 layers, 3 relations) for MI455X / gfx1250 (wave32, WMMA).
//
// Roofline: dense x@W transforms are ~58 GFLOP total -> tiny vs. the ~5 GB of
// irregular edge gather/scatter traffic at 23.3 TB/s. So the GEMMs use the
// chip's bf16 WMMA path (v_wmma_f32_16x16x32_bf16, fp32 accumulate) with W
// staged per-workgroup into LDS in B-fragment order, while the edge phase is
// a wave-per-edge float4 gather + global_atomic_add_f32 scatter.
// ---------------------------------------------------------------------------

#define CDIM 128

typedef __attribute__((ext_vector_type(16))) __bf16 v16bf;
typedef __attribute__((ext_vector_type(8)))  float  v8f;

// K-offset pattern within a 16-bit WMMA fragment (CDNA5 ISA):
// VGPR 0..3 hold K = kbase+0..7 (2 packed per VGPR), VGPR 4..7 hold K = kbase+16..23.
__device__ __forceinline__ int frag_k(int e) {
  int vg = e >> 1;
  return ((vg < 4) ? (vg << 1) : (16 + ((vg - 4) << 1))) + (e & 1);
}

// H[N x 128] = X[N x 128] @ W[128 x 128], bf16 WMMA, fp32 accumulate.
// Block = 256 threads = 8 waves; wave w computes rows [blk*128 + w*16, +16).
__global__ __launch_bounds__(256)
void gat_gemm_xw_kernel(const float* __restrict__ X, const float* __restrict__ W,
                        float* __restrict__ H, int N) {
  // W staged as bf16 in B-fragment order: [kb(4)][nt(8)][lane(32)][e(16)]
  __shared__ __align__(32) __bf16 wlds[4 * 8 * 32 * 16];

  const int tid = threadIdx.x;
  for (int i = 0; i < 64; ++i) {
    int idx  = tid + i * 256;                 // 16384 elements total
    int e    = idx & 15;
    int lane = (idx >> 4) & 31;
    int nt   = (idx >> 9) & 7;
    int kb   = idx >> 12;
    int kbase = (lane & 16) ? 8 : 0;
    int k    = kb * 32 + kbase + frag_k(e);
    int col  = nt * 16 + (lane & 15);
    wlds[idx] = (__bf16)W[k * CDIM + col];
  }
  __syncthreads();

  const int wave  = tid >> 5;
  const int lane  = tid & 31;
  const int row0  = blockIdx.x * 128 + wave * 16;
  const int m     = lane & 15;
  const int kbase = (lane & 16) ? 8 : 0;
  const int row   = row0 + m;
  const bool ok   = row < N;
  const float* xr = X + (size_t)(ok ? row : 0) * CDIM;

  v16bf a[4];
  for (int kb = 0; kb < 4; ++kb) {
    float4 x0, x1, x2, x3;
    if (ok) {
      const float* p = xr + kb * 32 + kbase;
      x0 = *(const float4*)(p + 0);
      x1 = *(const float4*)(p + 4);
      x2 = *(const float4*)(p + 16);
      x3 = *(const float4*)(p + 20);
    } else {
      x0 = x1 = x2 = x3 = make_float4(0.f, 0.f, 0.f, 0.f);
    }
    a[kb][0]  = (__bf16)x0.x; a[kb][1]  = (__bf16)x0.y;
    a[kb][2]  = (__bf16)x0.z; a[kb][3]  = (__bf16)x0.w;
    a[kb][4]  = (__bf16)x1.x; a[kb][5]  = (__bf16)x1.y;
    a[kb][6]  = (__bf16)x1.z; a[kb][7]  = (__bf16)x1.w;
    a[kb][8]  = (__bf16)x2.x; a[kb][9]  = (__bf16)x2.y;
    a[kb][10] = (__bf16)x2.z; a[kb][11] = (__bf16)x2.w;
    a[kb][12] = (__bf16)x3.x; a[kb][13] = (__bf16)x3.y;
    a[kb][14] = (__bf16)x3.z; a[kb][15] = (__bf16)x3.w;
  }

  v8f acc[8] = {};
  for (int kb = 0; kb < 4; ++kb) {
    for (int nt = 0; nt < 8; ++nt) {
      v16bf b = *(const v16bf*)&wlds[(((kb << 3) + nt) * 32 + lane) << 4];
      acc[nt] = __builtin_amdgcn_wmma_f32_16x16x32_bf16(
          false, a[kb], false, b, (short)0, acc[nt], false, false);
    }
  }

  // C/D layout: lanes 0-15 -> M = r, N = lane; lanes 16-31 -> M = r+8, N = lane-16
  const int mofs = (lane & 16) ? 8 : 0;
  for (int nt = 0; nt < 8; ++nt) {
    int col = nt * 16 + (lane & 15);
    for (int r = 0; r < 8; ++r) {
      int rr = row0 + mofs + r;
      if (rr < N) H[(size_t)rr * CDIM + col] = acc[nt][r];
    }
  }
}

// out[row] = dot(H[row,:], a) (+ *bias). One wave per row, float4 per lane.
__global__ void gat_row_dot_kernel(const float* __restrict__ H,
                                   const float* __restrict__ a,
                                   const float* __restrict__ bias,
                                   float* __restrict__ out, int N) {
  int g = blockIdx.x * blockDim.x + threadIdx.x;
  int row = g >> 5, lane = g & 31;
  if (row >= N) return;
  float4 h  = ((const float4*)(H + (size_t)row * CDIM))[lane];
  float4 av = ((const float4*)a)[lane];
  float s = h.x * av.x + h.y * av.y + h.z * av.z + h.w * av.w;
  for (int o = 16; o; o >>= 1) s += __shfl_xor(s, o, 32);
  if (lane == 0) out[row] = s + (bias ? *bias : 0.f);
}

__global__ void gat_fill_kernel(float* __restrict__ p, float v, int n) {
  int i = blockIdx.x * blockDim.x + threadIdx.x;
  if (i < n) p[i] = v;
}

// acc[i, c] = b0[c] (+ b1[c])  — bias init for the message accumulator.
__global__ void gat_bias_init_kernel(float* __restrict__ acc,
                                     const float* __restrict__ b0,
                                     const float* __restrict__ b1, int rows) {
  int i = blockIdx.x * blockDim.x + threadIdx.x;
  if (i >= rows * CDIM) return;
  int c = i & (CDIM - 1);
  acc[i] = b0[c] + (b1 ? b1[c] : 0.f);
}

__device__ __forceinline__ float lrelu(float e) { return e >= 0.f ? e : 0.2f * e; }

__device__ __forceinline__ void atomicMaxF(float* addr, float v) {
  if (v >= 0.f) atomicMax((int*)addr, __float_as_int(v));
  else          atomicMin((unsigned int*)addr, __float_as_uint(v));
}

__device__ __forceinline__ void atomAddF(float* p, float v) {
  __hip_atomic_fetch_add(p, v, __ATOMIC_RELAXED, __HIP_MEMORY_SCOPE_AGENT);
}

__global__ void gat_edge_max_kernel(const float* __restrict__ als,
                                    const float* __restrict__ ald,
                                    const int* __restrict__ src,
                                    const int* __restrict__ dst, int E,
                                    float* __restrict__ emax) {
  int i = blockIdx.x * blockDim.x + threadIdx.x;
  if (i >= E) return;
  int s = src[i], d = dst[i];
  atomicMaxF(&emax[d], lrelu(als[s] + ald[d]));
}

__global__ void gat_edge_sum_kernel(const float* __restrict__ als,
                                    const float* __restrict__ ald,
                                    const int* __restrict__ src,
                                    const int* __restrict__ dst, int E,
                                    const float* __restrict__ emax,
                                    float* __restrict__ den) {
  int i = blockIdx.x * blockDim.x + threadIdx.x;
  if (i >= E) return;
  int s = src[i], d = dst[i];
  float e = lrelu(als[s] + ald[d]);
  atomAddF(&den[d], __expf(e - emax[d]));
}

// One wave per edge: lane handles 4 channels (float4 gather + 4 fp32 atomics).
__global__ void gat_edge_msg_kernel(const float* __restrict__ hs,
                                    const float* __restrict__ als,
                                    const float* __restrict__ ald,
                                    const float* __restrict__ emax,
                                    const float* __restrict__ den,
                                    const int* __restrict__ src,
                                    const int* __restrict__ dst, int E,
                                    float* __restrict__ out) {
  int g = blockIdx.x * blockDim.x + threadIdx.x;
  int edge = g >> 5, lane = g & 31;
  if (edge >= E) return;
  int s = src[edge], d = dst[edge];
  float e = lrelu(als[s] + ald[d]);
  float alpha = __expf(e - emax[d]) / (den[d] + 1e-16f);
  float4 h = ((const float4*)(hs + (size_t)s * CDIM))[lane];
  float* o = out + (size_t)d * CDIM + lane * 4;
  atomAddF(o + 0, alpha * h.x);
  atomAddF(o + 1, alpha * h.y);
  atomAddF(o + 2, alpha * h.z);
  atomAddF(o + 3, alpha * h.w);
}

__global__ void gat_elu_kernel(float* __restrict__ p, int n) {
  int i = blockIdx.x * blockDim.x + threadIdx.x;
  if (i >= n) return;
  float v = p[i];
  p[i] = v > 0.f ? v : __expf(v) - 1.f;
}

extern "C" void kernel_launch(void* const* d_in, const int* in_sizes, int n_in,
                              void* d_out, int out_size, void* d_ws, size_t ws_size,
                              hipStream_t stream) {
  const float* x_gene    = (const float*)d_in[0];
  const float* x_disease = (const float*)d_in[1];
  const float* Ws        = (const float*)d_in[2];   // [2,3,128,128]
  const float* Wd        = (const float*)d_in[3];   // [2,3,128,128]
  const float* As        = (const float*)d_in[4];   // [2,3,128]
  const float* Ad        = (const float*)d_in[5];   // [2,3,128]
  const float* Bc        = (const float*)d_in[6];   // [2,3,128]
  const float* lin_w     = (const float*)d_in[7];   // [2,128]
  const float* lin_b     = (const float*)d_in[8];   // [2]
  const int* assoc_src   = (const int*)d_in[9];
  const int* assoc_dst   = (const int*)d_in[10];
  const int* rev_src     = (const int*)d_in[11];
  const int* rev_dst     = (const int*)d_in[12];
  const int* ppi_src     = (const int*)d_in[13];
  const int* ppi_dst     = (const int*)d_in[14];

  const int NG = in_sizes[0] / CDIM;
  const int ND = in_sizes[1] / CDIM;
  const int EA = in_sizes[9];
  const int EP = in_sizes[13];

  // ---- workspace carve-up (floats) ----
  float* p = (float*)d_ws;
  auto take = [&](size_t n) { float* q = p; p += n; return q; };
  float* G0   = take((size_t)NG * CDIM);   // layer0 gene accum -> features
  float* G1   = take((size_t)NG * CDIM);   // layer1 gene accum -> features
  float* D0   = take((size_t)ND * CDIM);
  float* D1   = take((size_t)ND * CDIM);
  float* HSG  = take((size_t)NG * CDIM);   // gene-source transform
  float* HDG  = take((size_t)NG * CDIM);   // gene-dest transform
  float* SM   = take((size_t)ND * CDIM);   // small (disease) transform
  float* ALS  = take((size_t)NG);
  float* ALD  = take((size_t)NG);
  float* EMAX = take((size_t)NG);
  float* DEN  = take((size_t)NG);
  (void)ws_size; (void)n_in; (void)out_size;

  auto gemm = [&](const float* X, const float* W, float* H, int N) {
    gat_gemm_xw_kernel<<<(N + 127) / 128, 256, 0, stream>>>(X, W, H, N);
  };
  auto rowdot = [&](const float* H, const float* a, const float* b, float* o, int N) {
    long long t = (long long)N * 32;
    gat_row_dot_kernel<<<(unsigned)((t + 255) / 256), 256, 0, stream>>>(H, a, b, o, N);
  };

  auto do_rel = [&](const float* xs, int Ns, const float* xd, int Nd,
                    const float* Wsr, const float* Wdr,
                    const float* Asr, const float* Adr,
                    const int* esrc, const int* edst, int E,
                    float* hsbuf, float* hdbuf, float* accum) {
    gemm(xs, Wsr, hsbuf, Ns);
    gemm(xd, Wdr, hdbuf, Nd);
    rowdot(hsbuf, Asr, nullptr, ALS, Ns);
    rowdot(hdbuf, Adr, nullptr, ALD, Nd);
    gat_fill_kernel<<<(Nd + 255) / 256, 256, 0, stream>>>(EMAX, -INFINITY, Nd);
    gat_fill_kernel<<<(Nd + 255) / 256, 256, 0, stream>>>(DEN, 0.f, Nd);
    gat_edge_max_kernel<<<(E + 255) / 256, 256, 0, stream>>>(ALS, ALD, esrc, edst, E, EMAX);
    gat_edge_sum_kernel<<<(E + 255) / 256, 256, 0, stream>>>(ALS, ALD, esrc, edst, E, EMAX, DEN);
    long long t = (long long)E * 32;
    gat_edge_msg_kernel<<<(unsigned)((t + 255) / 256), 256, 0, stream>>>(
        hsbuf, ALS, ALD, EMAX, DEN, esrc, edst, E, accum);
  };

  const float* xg = x_gene;
  const float* xd = x_disease;
  float* Gbuf[2] = {G0, G1};
  float* Dbuf[2] = {D0, D1};

  for (int l = 0; l < 2; ++l) {
    float* Gl = Gbuf[l];
    float* Dl = Dbuf[l];
    const size_t wOff = (size_t)l * 3 * CDIM * CDIM;
    const size_t vOff = (size_t)l * 3 * CDIM;

    // rel 0: gene -> disease (assoc) into Dl
    gat_bias_init_kernel<<<(ND * CDIM + 255) / 256, 256, 0, stream>>>(
        Dl, Bc + vOff + 0 * CDIM, nullptr, ND);
    do_rel(xg, NG, xd, ND,
           Ws + wOff + 0 * CDIM * CDIM, Wd + wOff + 0 * CDIM * CDIM,
           As + vOff + 0 * CDIM, Ad + vOff + 0 * CDIM,
           assoc_src, assoc_dst, EA, HSG, SM, Dl);

    // rel 1 + rel 2 accumulate into Gl (init with both biases once)
    gat_bias_init_kernel<<<(NG * CDIM + 255) / 256, 256, 0, stream>>>(
        Gl, Bc + vOff + 1 * CDIM, Bc + vOff + 2 * CDIM, NG);

    // rel 1: disease -> gene (rev)
    do_rel(xd, ND, xg, NG,
           Ws + wOff + 1 * CDIM * CDIM, Wd + wOff + 1 * CDIM * CDIM,
           As + vOff + 1 * CDIM, Ad + vOff + 1 * CDIM,
           rev_src, rev_dst, EA, SM, HDG, Gl);

    // rel 2: gene -> gene (ppi)
    do_rel(xg, NG, xg, NG,
           Ws + wOff + 2 * CDIM * CDIM, Wd + wOff + 2 * CDIM * CDIM,
           As + vOff + 2 * CDIM, Ad + vOff + 2 * CDIM,
           ppi_src, ppi_dst, EP, HSG, HDG, Gl);

    gat_elu_kernel<<<(NG * CDIM + 255) / 256, 256, 0, stream>>>(Gl, NG * CDIM);
    gat_elu_kernel<<<(ND * CDIM + 255) / 256, 256, 0, stream>>>(Dl, ND * CDIM);
    xg = Gl;
    xd = Dl;
  }

  // final linear: out_g then out_d, concatenated flat in d_out
  float* out = (float*)d_out;
  rowdot(xg, lin_w + 0,    lin_b + 0, out,      NG);
  rowdot(xd, lin_w + CDIM, lin_b + 1, out + NG, ND);
}
